// GATConv_80762565034628
// MI455X (gfx1250) — compile-verified, matrix operand order checked by reading, not run on hardware
//
#include <hip/hip_runtime.h>
#include <hip/hip_bf16.h>

// Problem sizes (fixed by the reference)
#define NN   8192
#define FIN  512
#define DD   64
#define HH   8

typedef __attribute__((ext_vector_type(16))) _Float16 v16h;
typedef __attribute__((ext_vector_type(8)))  _Float16 v8h;
typedef __attribute__((ext_vector_type(4)))  _Float16 v4h;
typedef __attribute__((ext_vector_type(8)))  float    v8f;
typedef __attribute__((ext_vector_type(4)))  float    v4f;

#define L2E 1.4426950408889634f
#define NEGBIG (-9e15f)

// gfx1250 async global->LDS (ASYNCcnt-tracked) if the toolchain exposes it.
// Probe-discovered signature: (b128_t as(1)*, b128_t as(3)*, imm, imm) where
// b128_t is a GCC-style int4 vector (clang prints as(1) as "__device__").
#if __has_builtin(__builtin_amdgcn_global_load_async_to_lds_b128)
#define HAVE_ASYNC_LDS 1
typedef int b128_t __attribute__((vector_size(16)));
#define GADDR(p) ((__attribute__((address_space(1))) b128_t*)(p))
#define LADDR(p) ((__attribute__((address_space(3))) b128_t*)(p))
#else
#define HAVE_ASYNC_LDS 0
#endif

#if __has_builtin(__builtin_amdgcn_s_wait_asynccnt)
#define WAIT_ASYNC0() __builtin_amdgcn_s_wait_asynccnt(0)
#else
#define WAIT_ASYNC0() asm volatile("s_wait_asynccnt 0x0" ::: "memory")
#endif

// ---------------------------------------------------------------------------
// Phase 0a: x (f32, [N,FIN]) -> xh (f16, [N,FIN]); NT streaming read
// ---------------------------------------------------------------------------
__global__ __launch_bounds__(256) void k_cvt_x(const float* __restrict__ x,
                                               _Float16* __restrict__ xh) {
  int i = blockIdx.x * 256 + threadIdx.x;          // exactly N*FIN/4 threads
  v4f v = __builtin_nontemporal_load((const v4f*)x + i);
  v4h o;
  o[0] = (_Float16)v[0]; o[1] = (_Float16)v[1];
  o[2] = (_Float16)v[2]; o[3] = (_Float16)v[3];
  ((v4h*)xh)[i] = o;
}

// ---------------------------------------------------------------------------
// Phase 0b: W (f32, [H,FIN,DD]) -> WT (f16, [H,DD,FIN])  (transpose + convert)
// ---------------------------------------------------------------------------
__global__ __launch_bounds__(256) void k_cvt_w(const float* __restrict__ W,
                                               _Float16* __restrict__ WT) {
  int i = blockIdx.x * 256 + threadIdx.x;          // exactly H*DD*FIN threads
  int k = i & (FIN - 1);
  int d = (i >> 9) & (DD - 1);
  int h = i >> 15;
  WT[i] = (_Float16)__builtin_nontemporal_load(&W[(h * FIN + k) * DD + d]);
}

// ---------------------------------------------------------------------------
// Phase 1: Wh = x @ W[h] via WMMA f16 (f32 accumulate).
// Per wave: one 16x64 tile. Epilogue: stage f32 tile in LDS, compute
// f1 = Wh@a1, f2 = Wh@a2 in-register, store WhT as f16 [H][DD][NN]
// (transposed so phase-2 B fragments are contiguous 16B loads).
// ---------------------------------------------------------------------------
__global__ __launch_bounds__(256) void k_wh(const _Float16* __restrict__ xh,
                                            const _Float16* __restrict__ WT,
                                            const float* __restrict__ a1,
                                            const float* __restrict__ a2,
                                            _Float16* __restrict__ WhT,
                                            float* __restrict__ f1,
                                            float* __restrict__ f2) {
  __shared__ float ldsT[8][16][64];                // per-wave 16x64 f32 tile
  const int h    = blockIdx.x >> 6;
  const int ig   = blockIdx.x & 63;
  const int wave = threadIdx.x >> 5;
  const int lane = threadIdx.x & 31;
  const int ln   = lane & 15;
  const int hi   = lane >> 4;
  const int i0   = (ig * 8 + wave) * 16;
  const int koff = hi * 8;                          // ISA A/B lane K-offset

  const _Float16* xrow  = xh + (size_t)(i0 + ln) * FIN;
  const _Float16* wbase = WT + (size_t)h * DD * FIN;

  v8f c0 = {0,0,0,0,0,0,0,0};
  v8f c1 = {0,0,0,0,0,0,0,0};
  v8f c2 = {0,0,0,0,0,0,0,0};
  v8f c3 = {0,0,0,0,0,0,0,0};

#define WH_CHUNK(CREG, CIDX)                                                  \
  {                                                                           \
    const _Float16* wd = wbase + (size_t)((CIDX)*16 + ln) * FIN + k + koff;   \
    v8h blo = *(const v8h*)(wd);                                              \
    v8h bhi = *(const v8h*)(wd + 16);                                         \
    v16h B = __builtin_shufflevector(blo, bhi, 0,1,2,3,4,5,6,7,               \
                                     8,9,10,11,12,13,14,15);                  \
    CREG = __builtin_amdgcn_wmma_f32_16x16x32_f16(false, A, false, B,         \
                                                  (short)0, CREG, false, false); \
  }

  for (int k = 0; k < FIN; k += 32) {
    v8h alo = *(const v8h*)(xrow + k + koff);
    v8h ahi = *(const v8h*)(xrow + k + koff + 16);
    v16h A = __builtin_shufflevector(alo, ahi, 0,1,2,3,4,5,6,7,
                                     8,9,10,11,12,13,14,15);
    WH_CHUNK(c0, 0)
    WH_CHUNK(c1, 1)
    WH_CHUNK(c2, 2)
    WH_CHUNK(c3, 3)
  }
#undef WH_CHUNK

  // Stage the f32 tile: C layout -> row = r + 8*hi, col = chunk*16 + ln
#pragma unroll
  for (int r = 0; r < 8; ++r) {
    ldsT[wave][r + 8 * hi][ln]      = c0[r];
    ldsT[wave][r + 8 * hi][16 + ln] = c1[r];
    ldsT[wave][r + 8 * hi][32 + ln] = c2[r];
    ldsT[wave][r + 8 * hi][48 + ln] = c3[r];
  }
  // (same-wave DS ops are in order; no barrier needed)

  // f1/f2: lane handles row ln, d-range [hi*32, hi*32+32)
  float p1 = 0.f, p2 = 0.f;
#pragma unroll
  for (int d = 0; d < 32; ++d) {
    float v = ldsT[wave][ln][hi * 32 + d];
    p1 = fmaf(v, a1[h * DD + hi * 32 + d], p1);
    p2 = fmaf(v, a2[h * DD + hi * 32 + d], p2);
  }
  p1 += __shfl_xor(p1, 16, 32);
  p2 += __shfl_xor(p2, 16, 32);
  if (!hi) {
    f1[h * NN + i0 + ln] = p1;
    f2[h * NN + i0 + ln] = p2;
  }

  // Transposed f16 store: WhT[h][d][i0..i0+15] contiguous (2x 16B per d-row)
  for (int dd = lane; dd < 64; dd += 32) {
    v8h o0, o1;
#pragma unroll
    for (int r = 0; r < 8; ++r) o0[r] = (_Float16)ldsT[wave][r][dd];
#pragma unroll
    for (int r = 0; r < 8; ++r) o1[r] = (_Float16)ldsT[wave][8 + r][dd];
    _Float16* dst = WhT + ((size_t)(h * DD + dd)) * NN + i0;
    *(v8h*)dst       = o0;
    *(v8h*)(dst + 8) = o1;
  }
}

// ---------------------------------------------------------------------------
// Phase 2: masked attention + online softmax + P@Wh, single pass over adj.
// One wave owns a 16-row tile of one head; 8 waves/block share the head and
// double-buffer the 32x64 WhT j-tile (+ f2 slice) in LDS. Tile staging uses
// GLOBAL_LOAD_ASYNC_TO_LDS_B128 (ASYNCcnt) when available. adj is streamed
// with non-temporal loads (read-once, keep L2 for WhT).
// ---------------------------------------------------------------------------
__global__ __launch_bounds__(256) void k_att(const float* __restrict__ adj,
                                             const _Float16* __restrict__ WhT,
                                             const float* __restrict__ f1,
                                             const float* __restrict__ f2,
                                             float* __restrict__ out) {
  __shared__ __align__(16) _Float16 ldsB[2][64][40];  // 40-halves row pad
  __shared__ __align__(16) float    ldsF2[2][32];

  const int h    = blockIdx.x >> 6;
  const int ig   = blockIdx.x & 63;
  const int wave = threadIdx.x >> 5;
  const int lane = threadIdx.x & 31;
  const int ln   = lane & 15;
  const int hi   = lane >> 4;
  const int i0   = (ig * 8 + wave) * 16;
  const int row  = i0 + ln;
  const int koff = hi * 8;

  const float f1r = f1[h * NN + row];
  const float* adjrow   = adj + (size_t)row * NN;
  const _Float16* wht_h = WhT + (size_t)h * DD * NN;
  const float* f2h      = f2 + h * NN;

  // cooperative tile loader mapping
  const int t    = threadIdx.x;
  const int ld_d = t >> 2;           // 0..63
  const int ld_p = (t & 3) * 8;      // 0,8,16,24

  v8f c0 = {0,0,0,0,0,0,0,0};
  v8f c1 = {0,0,0,0,0,0,0,0};
  v8f c2 = {0,0,0,0,0,0,0,0};
  v8f c3 = {0,0,0,0,0,0,0,0};
  float m = NEGBIG, lsum = 0.f;
  float f2stage;

  // preload j-tile 0 into buffer 0
#if HAVE_ASYNC_LDS
  __builtin_amdgcn_global_load_async_to_lds_b128(
      GADDR(wht_h + (size_t)ld_d * NN + ld_p),
      LADDR(&ldsB[0][ld_d][ld_p]), 0, 0);
  if (t < 32) ldsF2[0][t] = f2h[t];
  WAIT_ASYNC0();
#else
  v8h stage = *(const v8h*)(wht_h + (size_t)ld_d * NN + ld_p);
  *(v8h*)&ldsB[0][ld_d][ld_p] = stage;
  if (t < 32) ldsF2[0][t] = f2h[t];
#endif
  __syncthreads();

#define ATT_CHUNK(CREG, CIDX)                                                 \
  {                                                                           \
    v8h blo = *(const v8h*)&ldsB[buf][(CIDX)*16 + ln][koff];                  \
    v8h bhi = *(const v8h*)&ldsB[buf][(CIDX)*16 + ln][koff + 16];             \
    v16h B = __builtin_shufflevector(blo, bhi, 0,1,2,3,4,5,6,7,               \
                                     8,9,10,11,12,13,14,15);                  \
    CREG = __builtin_amdgcn_wmma_f32_16x16x32_f16(false, A, false, B,         \
                                                  (short)0, CREG, false, false); \
  }

  for (int jt = 0; jt < NN / 32; ++jt) {
    const int j   = jt * 32;
    const int buf = jt & 1;

    // issue next-tile staging early (prev readers of buf^1 passed last barrier)
    if (jt + 1 < NN / 32) {
#if HAVE_ASYNC_LDS
      __builtin_amdgcn_global_load_async_to_lds_b128(
          GADDR(wht_h + (size_t)ld_d * NN + (j + 32) + ld_p),
          LADDR(&ldsB[buf ^ 1][ld_d][ld_p]), 0, 0);
#else
      stage = *(const v8h*)(wht_h + (size_t)ld_d * NN + (j + 32) + ld_p);
#endif
      if (t < 32) f2stage = f2h[j + 32 + t];
    }

    // adj row slices (the 256 MB HBM stream): 4x NT b128 per lane
    const float* ap = adjrow + j + koff;
    v4f g0 = __builtin_nontemporal_load((const v4f*)(ap));
    v4f g1 = __builtin_nontemporal_load((const v4f*)(ap + 4));
    v4f g2 = __builtin_nontemporal_load((const v4f*)(ap + 16));
    v4f g3 = __builtin_nontemporal_load((const v4f*)(ap + 20));
    // f2 slices from LDS (same k pattern)
    v4f q0 = *(const v4f*)&ldsF2[buf][koff];
    v4f q1 = *(const v4f*)&ldsF2[buf][koff + 4];
    v4f q2 = *(const v4f*)&ldsF2[buf][koff + 16];
    v4f q3 = *(const v4f*)&ldsF2[buf][koff + 20];

    float fa[16] = {g0[0],g0[1],g0[2],g0[3], g1[0],g1[1],g1[2],g1[3],
                    g2[0],g2[1],g2[2],g2[3], g3[0],g3[1],g3[2],g3[3]};
    float fq[16] = {q0[0],q0[1],q0[2],q0[3], q1[0],q1[1],q1[2],q1[3],
                    q2[0],q2[1],q2[2],q2[3], q3[0],q3[1],q3[2],q3[3]};
    float ev[16];
#pragma unroll
    for (int i = 0; i < 16; ++i) {
      float e0 = f1r + fq[i];
      float el = fmaxf(e0, 0.2f * e0);             // leaky_relu, alpha=0.2
      ev[i] = (fa[i] > 0.f) ? el : NEGBIG;          // mask (matches reference)
    }
    float mloc = ev[0];
#pragma unroll
    for (int i = 1; i < 16; ++i) mloc = fmaxf(mloc, ev[i]);
    mloc = fmaxf(mloc, __shfl_xor(mloc, 16, 32));   // pair lanes hold same row

    bool empty = __all(mloc <= -8.9e15f);           // fully-masked tile skip
    if (!empty) {                                   // wave-uniform branch
      float mnew  = fmaxf(m, mloc);
      float scale = exp2f((m - mnew) * L2E);
      m = mnew;
      v16h A;
      float ps = 0.f;
#pragma unroll
      for (int i = 0; i < 16; ++i) {
        float p = exp2f((ev[i] - mnew) * L2E);
        ps += p;
        A[i] = (_Float16)p;
      }
      lsum = lsum * scale + ps;
      if (!__all(scale == 1.0f)) {                  // wave-uniform rescale
#pragma unroll
        for (int r = 0; r < 8; ++r) {
          float s = __shfl(scale, r + hi * 8, 32);  // row r+8*hi scale
          c0[r] = c0[r] * s; c1[r] = c1[r] * s;
          c2[r] = c2[r] * s; c3[r] = c3[r] * s;
        }
      }
      ATT_CHUNK(c0, 0)
      ATT_CHUNK(c1, 1)
      ATT_CHUNK(c2, 2)
      ATT_CHUNK(c3, 3)
    }

    // commit staged data; async tile must land before the barrier
    if (jt + 1 < NN / 32) {
#if !HAVE_ASYNC_LDS
      *(v8h*)&ldsB[buf ^ 1][ld_d][ld_p] = stage;
#endif
      if (t < 32) ldsF2[buf ^ 1][t] = f2stage;
    }
#if HAVE_ASYNC_LDS
    WAIT_ASYNC0();
#endif
    __syncthreads();
  }
#undef ATT_CHUNK

  // finalize: combine pair-lane partial sums, normalize, NT store
  float lt   = lsum + __shfl_xor(lsum, 16, 32);
  float rinv = 1.0f / lt;                           // self-loop => lt > 0
#pragma unroll
  for (int r = 0; r < 8; ++r) {
    float s = __shfl(rinv, r + hi * 8, 32);
    float* o = out + (size_t)(i0 + hi * 8 + r) * (HH * DD) + h * DD + ln;
    __builtin_nontemporal_store(c0[r] * s, o);
    __builtin_nontemporal_store(c1[r] * s, o + 16);
    __builtin_nontemporal_store(c2[r] * s, o + 32);
    __builtin_nontemporal_store(c3[r] * s, o + 48);
  }
}

// ---------------------------------------------------------------------------
// Host-side launcher
// ---------------------------------------------------------------------------
extern "C" void kernel_launch(void* const* d_in, const int* in_sizes, int n_in,
                              void* d_out, int out_size, void* d_ws, size_t ws_size,
                              hipStream_t stream) {
  const float* x   = (const float*)d_in[0];   // [N, FIN]
  const float* adj = (const float*)d_in[1];   // [N, N]
  const float* W   = (const float*)d_in[2];   // [H, FIN, DD]
  const float* a1  = (const float*)d_in[3];   // [H, DD]
  const float* a2  = (const float*)d_in[4];   // [H, DD]
  float* out = (float*)d_out;                 // [N, H*DD]

  char* ws = (char*)d_ws;
  const size_t XH_B  = (size_t)NN * FIN * 2;         // 8 MB
  const size_t WT_B  = (size_t)HH * DD * FIN * 2;    // 512 KB
  const size_t WHT_B = (size_t)HH * DD * NN * 2;     // 8 MB
  const size_t F_B   = (size_t)HH * NN * 4;          // 256 KB each
  _Float16* xh  = (_Float16*)(ws);
  _Float16* WT  = (_Float16*)(ws + XH_B);
  _Float16* WhT = (_Float16*)(ws + XH_B + WT_B);
  float*    f1  = (float*)(ws + XH_B + WT_B + WHT_B);
  float*    f2  = (float*)(ws + XH_B + WT_B + WHT_B + F_B);

  k_cvt_x<<<(NN * FIN / 4) / 256, 256, 0, stream>>>(x, xh);
  k_cvt_w<<<(HH * DD * FIN) / 256, 256, 0, stream>>>(W, WT);
  k_wh  <<<512, 256, 0, stream>>>(xh, WT, a1, a2, WhT, f1, f2);
  k_att <<<512, 256, 0, stream>>>(adj, WhT, f1, f2, out);
}
